// SlidingWindowAttention_19670950216037
// MI455X (gfx1250) — matrix operand order repaired and checked by reading, not compiled, not run on hardware
//
#include <hip/hip_runtime.h>
#include <hip/hip_bf16.h>

// ---------------------------------------------------------------------------
// SlidingWindowAttention for MI455X (gfx1250, wave32, WMMA bf16 16x16x32)
// B=2, T=2048, D=1024, H=16, HEAD_DIM=64, window=256
// GEMMs: async-LDS double-buffered B tiles, raw split barriers (no counter
// drain), branch-free register-pipelined A fragments, 32x64 wave tiles.
// ---------------------------------------------------------------------------

typedef __attribute__((ext_vector_type(16))) __bf16 v16bf;
typedef __attribute__((ext_vector_type(8)))  __bf16 v8bf;
typedef __attribute__((ext_vector_type(8)))  float  v8f;

union Frag16 { v16bf v; v8bf h[2]; };

#define WMMA_BF16(a, b, c) \
  __builtin_amdgcn_wmma_f32_16x16x32_bf16(false, (a), false, (b), (short)0, (c), false, false)

#ifndef DEV_INLINE
#define DEV_INLINE static __device__ __forceinline__
#endif
DEV_INLINE int imin(int a, int b) { return a < b ? a : b; }
DEV_INLINE int imax(int a, int b) { return a > b ? a : b; }

#if defined(__gfx1250__) && __has_builtin(__builtin_amdgcn_global_load_async_to_lds_b128)
#define HAVE_ASYNC_LDS 1
typedef int v4i_gcc __attribute__((__vector_size__(16)));
typedef __attribute__((address_space(1))) v4i_gcc gas_v4i;
typedef __attribute__((address_space(3))) v4i_gcc las_v4i;
#else
#define HAVE_ASYNC_LDS 0
#endif

// Raw workgroup split barrier: does NOT drain vmem counters (unlike
// __syncthreads), so register prefetch loads stay in flight across it.
DEV_INLINE void block_barrier() {
  asm volatile("s_barrier_signal -1\n\ts_barrier_wait -1" ::: "memory");
}

// Problem constants
static constexpr int Bb = 2;
static constexpr int Ts = 2048;
static constexpr int Dd = 1024;
static constexpr int Hh = 16;
static constexpr int HD = 64;
static constexpr int Mrows = Bb * Ts;       // 4096

// ---------------------------------------------------------------------------
// Conversion kernels
// ---------------------------------------------------------------------------
__global__ void k_cvt_bf16(const float* __restrict__ in, __bf16* __restrict__ out, int n) {
  int i = blockIdx.x * blockDim.x + threadIdx.x;
  if (i < n) out[i] = (__bf16)in[i];
}

// in: [K, N] f32 row-major  ->  out: [N, K] bf16 row-major (transpose)
__global__ void k_cvt_transpose(const float* __restrict__ in, __bf16* __restrict__ out,
                                int K, int N) {
  int i = blockIdx.x * blockDim.x + threadIdx.x;
  if (i < K * N) {
    int n = i / K, k = i - n * K;
    out[i] = (__bf16)in[(size_t)k * N + n];
  }
}

// ---------------------------------------------------------------------------
// Shared GEMM core: block tile 256x64 (8 waves x 32 rows), K-step 32.
// B tile (64 cols x 32 k = 4 KB) staged in LDS via async DMA, double buffered.
// Each wave issues ONE async b128 load covering 512 B; 8 waves fill the tile.
// Pipeline is branch-free: the last iteration stages a dummy tile (K offset
// wrapped to 0, lands in the already-consumed buffer; kernel-end wait-idle
// retires it), which keeps every fragment unconditionally defined and avoids
// register-allocator merge copies on the loop back-edge.
// ---------------------------------------------------------------------------
DEV_INLINE void stage_b_tile(const __bf16* __restrict__ WT, __bf16* dst,
                             int n0, int k0, int wave, int lane) {
  const int t = wave * 32 + lane;      // 0..255
  const int n = t >> 2, c = t & 3;     // row 0..63, 16B chunk 0..3
  const __bf16* gp = WT + (size_t)(n0 + n) * Dd + k0 + c * 8;
  __bf16* lp = dst + n * 32 + c * 8;   // LDS row stride = 32 elems (64 B)
#if HAVE_ASYNC_LDS
  __builtin_amdgcn_global_load_async_to_lds_b128((gas_v4i*)gp, (las_v4i*)lp, 0, 0);
#else
  *(v8bf*)lp = *(const v8bf*)gp;
#endif
}

DEV_INLINE void wait_b_stage() {
#if HAVE_ASYNC_LDS
  asm volatile("s_wait_asynccnt 0x1" ::: "memory");
#endif
}

DEV_INLINE Frag16 read_b_lds(const __bf16* buf, int nt, int mrow, int g) {
  Frag16 b;
  const __bf16* p = buf + (nt * 16 + mrow) * 32 + g * 8;
  b.h[0] = *(const v8bf*)(p);
  b.h[1] = *(const v8bf*)(p + 16);
  return b;
}

DEV_INLINE void load_a_pair(const __bf16* xrow0, const __bf16* xrow1,
                            int koff, int g, Frag16 a[2]) {
  a[0].h[0] = *(const v8bf*)(xrow0 + koff + g * 8);
  a[0].h[1] = *(const v8bf*)(xrow0 + koff + 16 + g * 8);
  a[1].h[0] = *(const v8bf*)(xrow1 + koff + g * 8);
  a[1].h[1] = *(const v8bf*)(xrow1 + koff + 16 + g * 8);
}

DEV_INLINE void gemm_core(const __bf16* __restrict__ X, const __bf16* __restrict__ WT,
                          __bf16* bsh, v8f acc[2][4], int m0, int n0, int wave, int lane) {
  const int mrow = lane & 15, g = lane >> 4;
  const __bf16* xrow0 = X + (size_t)(m0 + mrow) * Dd;        // rows m0..m0+15
  const __bf16* xrow1 = xrow0 + (size_t)16 * Dd;             // rows m0+16..m0+31
  constexpr int nIter = Dd / 32;       // 32 K-steps

  // prologue: stage first B tile, load first A fragments
  stage_b_tile(WT, bsh, n0, 0, wave, lane);
  Frag16 a_cur[2];
  load_a_pair(xrow0, xrow1, 0, g, a_cur);

  for (int it = 0; it < nIter; ++it) {
    const int kn = ((it + 1) * 32) & (Dd - 1);   // wraps to 0 on last iter
    __bf16* cur = bsh + (it & 1) * (64 * 32);
    __bf16* nxt = bsh + ((it + 1) & 1) * (64 * 32);

    stage_b_tile(WT, nxt, n0, kn, wave, lane);   // DMA next B tile (dummy on last)
    Frag16 a_nxt[2];
    load_a_pair(xrow0, xrow1, kn, g, a_nxt);     // prefetch next A (dummy on last)

    wait_b_stage();       // my portion of `cur` landed (next tile stays in flight)
    block_barrier();      // all waves' portions landed

#pragma unroll
    for (int nt = 0; nt < 4; ++nt) {
      Frag16 b = read_b_lds(cur, nt, mrow, g);
      acc[0][nt] = WMMA_BF16(a_cur[0].v, b.v, acc[0][nt]);
      acc[1][nt] = WMMA_BF16(a_cur[1].v, b.v, acc[1][nt]);
    }

    block_barrier();      // all reads of `cur` done before it is overwritten
    a_cur[0] = a_nxt[0];
    a_cur[1] = a_nxt[1];
  }
}

// ---------------------------------------------------------------------------
// GEMM 1: qkv = x @ w_qkv + b_qkv ; scatter to Q[B,H,T,64], K[B,H,T,64],
// Vt[B,H,64,T] (V stored transposed so P@V B-fragments load contiguously).
// ---------------------------------------------------------------------------
__global__ void __launch_bounds__(256)
k_gemm_qkv(const __bf16* __restrict__ X,    // [4096, 1024]
           const __bf16* __restrict__ WT,   // [3072, 1024]  (w_qkv^T)
           const float*  __restrict__ bias, // [3072]
           __bf16* __restrict__ Qb, __bf16* __restrict__ Kb, __bf16* __restrict__ Vtb) {
  __shared__ __align__(16) __bf16 bsh[2][64 * 32];    // 8 KB
  const int wave = threadIdx.x >> 5, lane = threadIdx.x & 31;
  const int mrow = lane & 15, g = lane >> 4;
  const int m0 = blockIdx.x * 256 + wave * 32;
  const int n0 = blockIdx.y * 64;

  v8f acc[2][4] = {};
  gemm_core(X, WT, &bsh[0][0], acc, m0, n0, wave, lane);

#pragma unroll
  for (int nt = 0; nt < 4; ++nt) {
    const int col = n0 + nt * 16 + mrow;          // 0..3071
    const float bv = bias[col];
    const int which = col >> 10;                  // 0=q 1=k 2=v
    const int rem = col & 1023;
    const int h = rem >> 6, d = rem & 63;
#pragma unroll
    for (int mi = 0; mi < 2; ++mi) {
#pragma unroll
      for (int i = 0; i < 8; ++i) {
        const int row = m0 + mi * 16 + i + 8 * g; // b*T + t
        const int bb = row >> 11, t = row & (Ts - 1);
        const float val = acc[mi][nt][i] + bv;
        const size_t bh = (size_t)bb * Hh + h;
        if (which == 0)      Qb[(bh * Ts + t) * HD + d]  = (__bf16)val;
        else if (which == 1) Kb[(bh * Ts + t) * HD + d]  = (__bf16)val;
        else                 Vtb[(bh * HD + d) * Ts + t] = (__bf16)val;
      }
    }
  }
}

// ---------------------------------------------------------------------------
// Attention: one wave per 16-query tile; online softmax over 32-key blocks.
// ---------------------------------------------------------------------------
__global__ void __launch_bounds__(256)
k_attn(const __bf16* __restrict__ Qb, const __bf16* __restrict__ Kb,
       const __bf16* __restrict__ Vtb, __bf16* __restrict__ attnb,
       const int* __restrict__ win) {
  __shared__ __align__(16) __bf16 plds[8][16 * 32];   // 1 KB per wave

  const int W = *win;
  const int wave = threadIdx.x >> 5, lane = threadIdx.x & 31;
  const int mrow = lane & 15, g = lane >> 4;

  const int gid = blockIdx.x * 8 + wave;   // 0..4095
  const int qt = gid & (Ts / 16 - 1);      // 128 query tiles per (b,h)
  const int bh = gid >> 7;                 // 0..31
  const int q0 = qt * 16;

  const __bf16* Q  = Qb  + (size_t)bh * Ts * HD;
  const __bf16* K  = Kb  + (size_t)bh * Ts * HD;
  const __bf16* Vt = Vtb + (size_t)bh * HD * Ts;

  // Q A-fragments for head-dim 0..31 and 32..63
  Frag16 aq0, aq1;
  {
    const __bf16* qrow = Q + (size_t)(q0 + mrow) * HD;
    aq0.h[0] = *(const v8bf*)(qrow + g * 8);
    aq0.h[1] = *(const v8bf*)(qrow + 16 + g * 8);
    aq1.h[0] = *(const v8bf*)(qrow + 32 + g * 8);
    aq1.h[1] = *(const v8bf*)(qrow + 48 + g * 8);
  }

  v8f o[4] = {};
  float rm[8], rl[8];
#pragma unroll
  for (int i = 0; i < 8; ++i) { rm[i] = -1e30f; rl[i] = 0.f; }

  const float scale = 0.125f;   // 1/sqrt(64)
  const int ks = imax(0, q0 - W) & ~31;

  for (int k0 = ks; k0 <= q0 + 15; k0 += 32) {
    // ---- scores: two 16x16 key tiles, contraction over head dim (2 WMMA each)
    v8f s0 = {}, s1 = {};
    {
      const __bf16* krow = K + (size_t)(k0 + mrow) * HD;
      Frag16 b0, b1;
      b0.h[0] = *(const v8bf*)(krow + g * 8);
      b0.h[1] = *(const v8bf*)(krow + 16 + g * 8);
      b1.h[0] = *(const v8bf*)(krow + 32 + g * 8);
      b1.h[1] = *(const v8bf*)(krow + 48 + g * 8);
      s0 = WMMA_BF16(aq0.v, b0.v, s0);
      s0 = WMMA_BF16(aq1.v, b1.v, s0);
    }
    {
      const int kk = imin(k0 + 16 + mrow, Ts - 1);
      const __bf16* krow = K + (size_t)kk * HD;
      Frag16 b0, b1;
      b0.h[0] = *(const v8bf*)(krow + g * 8);
      b0.h[1] = *(const v8bf*)(krow + 16 + g * 8);
      b1.h[0] = *(const v8bf*)(krow + 32 + g * 8);
      b1.h[1] = *(const v8bf*)(krow + 48 + g * 8);
      s1 = WMMA_BF16(aq0.v, b0.v, s1);
      s1 = WMMA_BF16(aq1.v, b1.v, s1);
    }

    // prefetch next key block's K rows (global_prefetch_b8)
    if (k0 + 32 <= q0 + 15) {
      __builtin_prefetch(K + (size_t)(k0 + 32 + mrow) * HD, 0, 1);
      __builtin_prefetch(K + (size_t)(k0 + 48 + mrow) * HD, 0, 1);
    }

    // ---- scale, mask, online-softmax update (rows live across 16-lane groups)
#pragma unroll
    for (int i = 0; i < 8; ++i) {
      const int q = q0 + i + 8 * g;
      const int key0 = k0 + mrow;
      const int key1 = key0 + 16;
      float v0 = s0[i] * scale, v1 = s1[i] * scale;
      const int d0 = q - key0, d1 = q - key1;
      if (!(d0 >= 0 && d0 <= W)) v0 = -1e30f;
      if (!(d1 >= 0 && d1 <= W) || key1 >= Ts) v1 = -1e30f;

      float mx = fmaxf(v0, v1);
#pragma unroll
      for (int off = 1; off < 16; off <<= 1) mx = fmaxf(mx, __shfl_xor(mx, off, 32));
      const float mnew = fmaxf(rm[i], mx);
      const float a = __expf(rm[i] - mnew);
      const float p0 = __expf(v0 - mnew);
      const float p1 = __expf(v1 - mnew);
      float ts = p0 + p1;
#pragma unroll
      for (int off = 1; off < 16; off <<= 1) ts += __shfl_xor(ts, off, 32);
      rl[i] = rl[i] * a + ts;
      rm[i] = mnew;
      s0[i] = p0; s1[i] = p1;
      o[0][i] *= a; o[1][i] *= a; o[2][i] *= a; o[3][i] *= a;
    }

    // ---- V fragments issued first so global loads overlap the LDS round-trip
    Frag16 vb[4];
#pragma unroll
    for (int nt = 0; nt < 4; ++nt) {
      const int d = nt * 16 + mrow;
      const int tb = imin(k0 + g * 16, Ts - 16);
      const __bf16* vrow = Vt + (size_t)d * Ts + tb;
      vb[nt].h[0] = *(const v8bf*)(vrow);
      vb[nt].h[1] = *(const v8bf*)(vrow + 8);
    }

    // ---- P: C-fragment -> A-fragment via per-wave LDS (bf16)
    __bf16* pl = &plds[wave][0];
#pragma unroll
    for (int i = 0; i < 8; ++i) {
      const int row = i + 8 * g;
      pl[row * 32 + mrow]      = (__bf16)s0[i];
      pl[row * 32 + 16 + mrow] = (__bf16)s1[i];
    }
    asm volatile("s_wait_dscnt 0" ::: "memory");
    Frag16 pa;
    pa.h[0] = *(const v8bf*)(pl + mrow * 32 + g * 16);
    pa.h[1] = *(const v8bf*)(pl + mrow * 32 + g * 16 + 8);

    // ---- O += P @ V
#pragma unroll
    for (int nt = 0; nt < 4; ++nt) {
      o[nt] = WMMA_BF16(pa.v, vb[nt].v, o[nt]);
    }
  }

  // ---- finalize: divide by row sums, store bf16 [B*T, D] row-major
  const int b = bh >> 4, h = bh & 15;
#pragma unroll
  for (int i = 0; i < 8; ++i) {
    const float inv = 1.0f / rl[i];
    const int t = q0 + i + 8 * g;
    const size_t rowbase = ((size_t)b * Ts + t) * Dd + h * HD;
#pragma unroll
    for (int nt = 0; nt < 4; ++nt) {
      attnb[rowbase + nt * 16 + mrow] = (__bf16)(o[nt][i] * inv);
    }
  }
}

// ---------------------------------------------------------------------------
// GEMM 2: out = attn @ w_o + b_o  (f32 output)
// ---------------------------------------------------------------------------
__global__ void __launch_bounds__(256)
k_gemm_out(const __bf16* __restrict__ A,    // [4096, 1024]
           const __bf16* __restrict__ WT,   // [1024, 1024] (w_o^T)
           const float*  __restrict__ bias, // [1024]
           float* __restrict__ out) {       // [4096, 1024]
  __shared__ __align__(16) __bf16 bsh[2][64 * 32];    // 8 KB
  const int wave = threadIdx.x >> 5, lane = threadIdx.x & 31;
  const int mrow = lane & 15, g = lane >> 4;
  const int m0 = blockIdx.x * 256 + wave * 32;
  const int n0 = blockIdx.y * 64;

  v8f acc[2][4] = {};
  gemm_core(A, WT, &bsh[0][0], acc, m0, n0, wave, lane);

#pragma unroll
  for (int nt = 0; nt < 4; ++nt) {
    const int col = n0 + nt * 16 + mrow;
    const float bv = bias[col];
#pragma unroll
    for (int mi = 0; mi < 2; ++mi) {
#pragma unroll
      for (int i = 0; i < 8; ++i) {
        const int row = m0 + mi * 16 + i + 8 * g;
        out[(size_t)row * Dd + col] = acc[mi][nt][i] + bv;
      }
    }
  }
}

// ---------------------------------------------------------------------------
// Launch
// ---------------------------------------------------------------------------
extern "C" void kernel_launch(void* const* d_in, const int* in_sizes, int n_in,
                              void* d_out, int out_size, void* d_ws, size_t ws_size,
                              hipStream_t stream) {
  (void)in_sizes; (void)n_in; (void)out_size; (void)ws_size;
  const float* x     = (const float*)d_in[0];
  const float* w_qkv = (const float*)d_in[1];
  const float* b_qkv = (const float*)d_in[2];
  const float* w_o   = (const float*)d_in[3];
  const float* b_o   = (const float*)d_in[4];
  const int*   win   = (const int*)d_in[5];
  float* out = (float*)d_out;

  // Workspace layout (bf16 elements), total 48 MB
  __bf16* xb    = (__bf16*)d_ws;                         // 4096*1024
  __bf16* wqkvT = xb    + (size_t)Mrows * Dd;            // 3072*1024
  __bf16* woT   = wqkvT + (size_t)3 * Dd * Dd;           // 1024*1024
  __bf16* Qb    = woT   + (size_t)Dd * Dd;               // 32*2048*64
  __bf16* Kb    = Qb    + (size_t)Bb * Hh * Ts * HD;
  __bf16* Vtb   = Kb    + (size_t)Bb * Hh * Ts * HD;
  __bf16* attnb = Vtb   + (size_t)Bb * Hh * Ts * HD;     // 4096*1024

  const int nx = Mrows * Dd;             // 4194304
  const int nqkv = Dd * 3 * Dd;          // 3145728
  const int nwo = Dd * Dd;               // 1048576

  k_cvt_bf16<<<(nx + 255) / 256, 256, 0, stream>>>(x, xb, nx);
  k_cvt_transpose<<<(nqkv + 255) / 256, 256, 0, stream>>>(w_qkv, wqkvT, Dd, 3 * Dd);
  k_cvt_transpose<<<(nwo + 255) / 256, 256, 0, stream>>>(w_o, woT, Dd, Dd);

  k_gemm_qkv<<<dim3(Mrows / 256, (3 * Dd) / 64), 256, 0, stream>>>(
      xb, wqkvT, b_qkv, Qb, Kb, Vtb);

  k_attn<<<(Bb * Hh * (Ts / 16)) / 8, 256, 0, stream>>>(Qb, Kb, Vtb, attnb, win);

  k_gemm_out<<<dim3(Mrows / 256, Dd / 64), 256, 0, stream>>>(attnb, woT, b_o, out);
}
